// TTSDecoder_54425825575035
// MI455X (gfx1250) — matrix operand order repaired and checked by reading, not compiled, not run on hardware
//
#include <hip/hip_runtime.h>
#include <hip/hip_bf16.h>

// ---------------------------------------------------------------------------
// Types & helpers
// ---------------------------------------------------------------------------
typedef __attribute__((ext_vector_type(16))) __bf16 v16bf;
typedef __attribute__((ext_vector_type(8)))  float  v8f;
typedef __attribute__((ext_vector_type(4)))  unsigned int u32x4;
typedef __attribute__((ext_vector_type(8)))  int i32x8;
typedef __attribute__((ext_vector_type(4)))  int i32x4;

#if defined(__has_builtin)
#  if __has_builtin(__builtin_amdgcn_tensor_load_to_lds) && \
      __has_builtin(__builtin_amdgcn_s_wait_tensorcnt) && \
      __has_builtin(__builtin_amdgcn_groupstaticsize)
#    define HAVE_TDM 1
#  else
#    define HAVE_TDM 0
#  endif
#else
#  define HAVE_TDM 0
#endif

__device__ __forceinline__ __bf16 hbf(float f) { return (__bf16)f; }   // hw cvt

__device__ __forceinline__ v8f wmma_bf16(v16bf a, v16bf b, v8f c) {
  return __builtin_amdgcn_wmma_f32_16x16x32_bf16(false, a, false, b, (short)0, c,
                                                 false, false);
}

// WMMA A-fragment addressing (16x32 bf16 tile, ISA 7.12.2):
// lane<16: M=lane, K={0..7,16..23}; lane>=16: M=lane-16, K={8..15,24..31}.
__device__ __forceinline__ int fragIndex(int r, int k) {
  int kt = k >> 5, kk = k & 31;
  int lane = r + (((kk >> 3) & 1) << 4);
  int j = (kk & 7) + (((kk >> 4) & 1) << 3);
  return kt * 512 + lane * 16 + j;
}

// Gather A-fragment from fp32 row-major panel (used only in prenet GEMM).
__device__ __forceinline__ v16bf load_a_frag(const float* A, int lda, int kt, int lane) {
  const float* row = A + (size_t)(lane & 15) * lda;
  const int kb = kt * 32 + ((lane & 16) ? 8 : 0);
  v16bf a;
#pragma unroll
  for (int j = 0; j < 16; ++j) a[j] = hbf(row[kb + (j < 8 ? j : j + 8)]);
  return a;
}

__device__ __forceinline__ float sigmf(float x) { return 1.0f / (1.0f + __expf(-x)); }

#if HAVE_TDM
// One-shot TDM stage: contiguous global -> LDS copy of n8 8-byte units.
// D# per CDNA5 ISA ch.8: group0 {count=1, lds_addr, global_addr, type=2},
// group1 {data_size=8B, 1-D tensor: tensor_dim0 = tile_dim0 = n8}.
__device__ __forceinline__ void tdm_stage(const void* gsrc, unsigned ldsByteOff, int n8)
{
  unsigned long long ga = (unsigned long long)(size_t)gsrc;
  u32x4 g0 = { 1u,                                   // count=1, no gather
               ldsByteOff,                           // lds_addr
               (unsigned)(ga & 0xffffffffu),
               (unsigned)((ga >> 32) & 0x01ffffffu) | (2u << 30) };  // type=2
  i32x8 g1;
  g1[0] = 0x30000;                                   // data_size = 8B, mask=0
  g1[1] = (int)(((unsigned)n8 & 0xffffu) << 16);     // tensor_dim0 lo16
  g1[2] = (int)(((unsigned)n8 >> 16) | (1u << 16));  // tensor_dim0 hi16, tensor_dim1=1
  g1[3] = (int)(((unsigned)n8 & 0xffffu) << 16);     // tile_dim0
  g1[4] = 0;                                         // tile_dim1/2 unused
  g1[5] = n8;                                        // tensor_dim0_stride lo32
  g1[6] = 0;
  g1[7] = 0;
  i32x4 z4 = {0, 0, 0, 0};
#if __clang_major__ >= 23
  i32x8 z8 = {0, 0, 0, 0, 0, 0, 0, 0};
  __builtin_amdgcn_tensor_load_to_lds(g0, g1, z4, z4, z8, 0);
#else
  __builtin_amdgcn_tensor_load_to_lds(g0, g1, z4, z4, 0);
#endif
}
#endif

// ---------------------------------------------------------------------------
// Weight packing into WMMA B-fragment order (bf16).
// B tile (32Kx16N): lane<16 -> N=lane, K={0..7,16..23}; lane>=16 -> N=lane-16, K={8..15,24..31}.
// ---------------------------------------------------------------------------
__global__ void k_pack_b(const float* __restrict__ W, __bf16* __restrict__ out,
                         int Ntiles, int KtilesTotal, int kt0, int ktCount,
                         int strideK, int strideN, int srcOff,
                         int krowOff, int Ksrc, int Nsrc)
{
  int tid = blockIdx.x * blockDim.x + threadIdx.x;
  int total = ktCount * Ntiles * 32;
  if (tid >= total) return;
  int lane = tid & 31;
  int tile = tid >> 5;
  int ktl = tile % ktCount;
  int nt  = tile / ktCount;
  int kt  = kt0 + ktl;
  int n   = nt * 16 + (lane & 15);
  int kb  = kt * 32 + ((lane & 16) ? 8 : 0);
  __bf16* dst = out + ((size_t)nt * KtilesTotal + kt) * 512 + lane * 16;
  for (int j = 0; j < 16; ++j) {
    int k = kb + (j < 8 ? j : j + 8) - krowOff;
    float v = (k >= 0 && k < Ksrc && n < Nsrc)
                  ? W[(size_t)k * strideK + (size_t)n * strideN + srcOff]
                  : 0.0f;
    dst[j] = hbf(v);
  }
}

// ---------------------------------------------------------------------------
// Generic WMMA GEMM (prenet): C = act(A @ Bpacked + bias). One tile/wave.
// ---------------------------------------------------------------------------
__global__ void k_gemm(const float* __restrict__ A, const __bf16* __restrict__ Bp,
                       const float* __restrict__ bias, float* __restrict__ C,
                       int Mtiles, int Ntiles, int Ktiles, int lda, int N, int relu)
{
  int gw = (blockIdx.x * blockDim.x + threadIdx.x) >> 5;
  int lane = threadIdx.x & 31;
  if (gw >= Mtiles * Ntiles) return;
  int mt = gw / Ntiles, nt = gw % Ntiles;
  const float* At = A + (size_t)mt * 16 * lda;
  const __bf16* bp = Bp + (size_t)nt * Ktiles * 512 + lane * 16;
  v8f acc = {};
  for (int kt = 0; kt < Ktiles; ++kt) {
    __builtin_prefetch(bp + (size_t)(kt + 1) * 512, 0, 1);
    v16bf af = load_a_frag(At, lda, kt, lane);
    v16bf bf = *(const v16bf*)(bp + (size_t)kt * 512);
    acc = wmma_bf16(af, bf, acc);
  }
  int n  = nt * 16 + (lane & 15);
  int m0 = mt * 16 + ((lane & 16) ? 8 : 0);
  float bb = bias ? bias[n] : 0.0f;
#pragma unroll
  for (int r = 0; r < 8; ++r) {
    float v = acc[r] + bb;
    if (relu) v = fmaxf(v, 0.0f);
    C[(size_t)(m0 + r) * N + n] = v;
  }
}

// ---------------------------------------------------------------------------
// Prenet input: seq[(t*16+b), 0:96] = (t==0 ? go : x[b,:,t-1]), zero-padded K.
// ---------------------------------------------------------------------------
__global__ void k_build_seq(const float* __restrict__ x, const float* __restrict__ go,
                            float* __restrict__ seq)
{
  int i = blockIdx.x * blockDim.x + threadIdx.x;
  if (i >= 8192 * 96) return;
  int c = i % 96, r = i / 96;
  int t = r >> 4, b = r & 15;
  float v = 0.0f;
  if (c < 80) v = (t == 0) ? go[c] : x[((size_t)b * 80 + c) * 512 + (t - 1)];
  seq[i] = v;
}

__global__ void k_init(int* __restrict__ bar, float* __restrict__ alignOut, size_t n)
{
  size_t i = (size_t)blockIdx.x * blockDim.x + threadIdx.x;
  if (i < 4) bar[i] = 0;
  size_t stride = (size_t)gridDim.x * blockDim.x;
  for (size_t j = i; j < n; j += stride) alignOut[j] = 0.0f;
}

// ---------------------------------------------------------------------------
// Grid-wide sense-reversing barrier (16 resident blocks).
// ---------------------------------------------------------------------------
__device__ __forceinline__ void gsync(int* bar)
{
  __syncthreads();
  if (threadIdx.x == 0) {
    __threadfence();
    int gen = atomicAdd(&bar[1], 0);
    if (atomicAdd(&bar[0], 1) == (int)gridDim.x - 1) {
      atomicExch(&bar[0], 0);
      atomicAdd(&bar[1], 1);
    } else {
      while (atomicAdd(&bar[1], 0) == gen) { __builtin_amdgcn_s_sleep(2); }
    }
  }
  __syncthreads();
  __threadfence();
}

// ---------------------------------------------------------------------------
// Persistent recurrent decoder. 16 blocks x 512 threads (256 waves).
// Activations are produced directly in bf16 WMMA-fragment order (aFrag /
// qFrag / melFrag); GEMM phases stage aFrag to LDS via the Tensor Data Mover
// (fallback: uint4 copies) and issue WMMAs with dual accumulators.
// ---------------------------------------------------------------------------
struct DecArgs {
  const float *din, *memory;
  const int *memlen, *outlen;
  const __bf16 *AWp, *DWp, *WqP, *WmP;
  const float *ab, *db;
  const float *bq, *V, *Fw, *U, *Tw, *Tb, *vw, *bm;
  __bf16 *aFrag, *qFrag, *melFrag;
  float *gates, *ah, *ac, *dh, *dc, *pa, *ctx, *q16, *G;
  float *fbuf, *gbuf, *prior, *ebuf;
  float *melbuf, *outMel, *outAlign;
  int *bar;
  int maxin;
};

__device__ __forceinline__ void stageA(const __bf16* aFrag, __bf16* sAbf, int n16 /*uint4s*/)
{
#if HAVE_TDM
  if ((threadIdx.x >> 5) == 0) {
    tdm_stage(aFrag, __builtin_amdgcn_groupstaticsize(), n16 * 2 /*8B units*/);
    __builtin_amdgcn_s_wait_tensorcnt(0);
  }
  __syncthreads();
#else
  const uint4* s = (const uint4*)aFrag;
  uint4* d = (uint4*)sAbf;
  for (int i = threadIdx.x; i < n16; i += blockDim.x) d[i] = s[i];
  __syncthreads();
#endif
}

__global__ __launch_bounds__(512, 1) void k_decoder(DecArgs A)
{
  extern __shared__ __align__(16) unsigned char smem[];
  __bf16* sAbf = (__bf16*)smem;                 // up to 88 tiles * 512 bf16
  const int tid  = blockIdx.x * blockDim.x + threadIdx.x;
  const int nth  = gridDim.x * blockDim.x;      // 8192
  const int lane = threadIdx.x & 31;
  const int gw   = tid >> 5;                    // 0..255

  // Beta-binomial(10, 0.1, 0.9) pmf, reversed (attention prior).
  const float PR[11] = {0.013173f, 0.013028f, 0.013752f, 0.014978f, 0.016759f,
                        0.019322f, 0.023170f, 0.029470f, 0.041573f, 0.074749f,
                        0.740010f};

  // ---- initial state ----
  for (int i = tid; i < 16 * 1024; i += nth) { A.ah[i] = 0.f; A.ac[i] = 0.f; A.dh[i] = 0.f; A.dc[i] = 0.f; }
  for (int i = tid; i < 4096; i += nth) A.pa[i] = ((i & 255) == 0) ? 1.0f : 0.0f;
  for (int i = tid; i < 16 * 768; i += nth) A.ctx[i] = 0.0f;
  for (int i = tid; i < 16 * 2048; i += nth) {   // A1(t=0) = [xt | 0 | 0]
    int b = i >> 11, c = i & 2047;
    float v = (c < 256) ? A.din[(size_t)b * 256 + c] : 0.0f;
    A.aFrag[fragIndex(b, c)] = hbf(v);
  }
  gsync(A.bar);

  for (int t = 0; t < 512; ++t) {
    // ---- gate GEMM 1: A1=[xt|ctx|ah] (K=2048, 64 tiles) @ [Wi;Wh] ----
    stageA(A.aFrag, sAbf, 4096);
    {
      const __bf16* bp = A.AWp + (size_t)gw * 64 * 512 + lane * 16;
      const __bf16* ap = sAbf + lane * 16;
      v8f a0 = {}, a1 = {};
      for (int kt = 0; kt < 64; kt += 2) {
        __builtin_prefetch(bp + (size_t)(kt + 2) * 512, 0, 1);
        v16bf x0 = *(const v16bf*)(ap + (size_t)kt * 512);
        v16bf w0 = *(const v16bf*)(bp + (size_t)kt * 512);
        a0 = wmma_bf16(x0, w0, a0);
        v16bf x1 = *(const v16bf*)(ap + (size_t)(kt + 1) * 512);
        v16bf w1 = *(const v16bf*)(bp + (size_t)(kt + 1) * 512);
        a1 = wmma_bf16(x1, w1, a1);
      }
      v8f acc = a0 + a1;
      int n = gw * 16 + (lane & 15);
      int m0 = (lane & 16) ? 8 : 0;
      float bb = A.ab[n];
#pragma unroll
      for (int r = 0; r < 8; ++r) A.gates[(size_t)(m0 + r) * 4096 + n] = acc[r] + bb;
    }
    gsync(A.bar);

    // ---- attn LSTM pointwise + zoneout; emit qFrag (bf16 fragments of ah) ----
    for (int i = tid; i < 16 * 1024; i += nth) {
      int b = i >> 10, j = i & 1023;
      const float* g = A.gates + (size_t)b * 4096;
      float ig = sigmf(g[j]), fg = sigmf(g[1024 + j]);
      float gg = tanhf(g[2048 + j]), og = sigmf(g[3072 + j]);
      float cn = fg * A.ac[i] + ig * gg;
      float hn = og * tanhf(cn);
      float nh = 0.1f * A.ah[i] + 0.9f * hn;
      A.ah[i] = nh;
      A.ac[i] = 0.1f * A.ac[i] + 0.9f * cn;
      A.qFrag[fragIndex(b, j)] = hbf(nh);
    }
    gsync(A.bar);

    // ---- q16 = tanh(ah@Wq+bq) via WMMA (waves 0..7); static conv f; prior ----
    if (gw < 8) {
      const __bf16* bp = A.WqP + (size_t)gw * 32 * 512 + lane * 16;
      const __bf16* ap = A.qFrag + lane * 16;
      v8f a0 = {}, a1 = {};
      for (int kt = 0; kt < 32; kt += 2) {
        v16bf x0 = *(const v16bf*)(ap + (size_t)kt * 512);
        v16bf w0 = *(const v16bf*)(bp + (size_t)kt * 512);
        a0 = wmma_bf16(x0, w0, a0);
        v16bf x1 = *(const v16bf*)(ap + (size_t)(kt + 1) * 512);
        v16bf w1 = *(const v16bf*)(bp + (size_t)(kt + 1) * 512);
        a1 = wmma_bf16(x1, w1, a1);
      }
      v8f acc = a0 + a1;
      int n = gw * 16 + (lane & 15);
      int b0 = (lane & 16) ? 8 : 0;
      float bb = A.bq[n];
#pragma unroll
      for (int r = 0; r < 8; ++r) A.q16[(size_t)(b0 + r) * 128 + n] = tanhf(acc[r] + bb);
    }
    for (int i = tid; i < 32768; i += nth) {            // f[b,t,c]
      int c = i & 7, tt = (i >> 3) & 255, b = i >> 11;
      const float* par = A.pa + b * 256;
      const float* fw = A.Fw + c * 21;
      float s = 0.0f;
      for (int k = 0; k < 21; ++k) {
        int ts = tt + k - 10;
        if (ts >= 0 && ts < 256) s += fw[k] * par[ts];
      }
      A.fbuf[i] = s;
    }
    for (int i = tid; i < 4096; i += nth) {             // prior[b,t]
      int b = i >> 8, tt = i & 255;
      const float* par = A.pa + b * 256;
      float s = 0.0f;
      for (int j = 0; j < 11; ++j) {
        int ts = tt + j - 10;
        if (ts >= 0) s += PR[j] * par[ts];
      }
      A.prior[i] = __logf(fmaxf(s, 1e-6f));
    }
    gsync(A.bar);

    // ---- G = q16 @ V (16x168, K=128) ----
    for (int i = tid; i < 16 * 168; i += nth) {
      int b = i / 168, j = i % 168;
      const float* q = A.q16 + b * 128;
      float s = 0.0f;
      for (int d = 0; d < 128; ++d) s += q[d] * A.V[(size_t)d * 168 + j];
      A.G[i] = s;
    }
    gsync(A.bar);

    // ---- dynamic conv g[b,t,c] ----
    for (int i = tid; i < 32768; i += nth) {
      int c = i & 7, tt = (i >> 3) & 255, b = i >> 11;
      const float* par = A.pa + b * 256;
      const float* fl = A.G + b * 168 + c * 21;
      float s = 0.0f;
      for (int k = 0; k < 21; ++k) {
        int ts = tt + k - 10;
        if (ts >= 0 && ts < 256) s += fl[k] * par[ts];
      }
      A.gbuf[i] = s;
    }
    gsync(A.bar);

    // ---- energies e[b,t] = vw . tanh(f@U + g@Tw + Tb) + prior, masked ----
    for (int i = tid; i < 4096; i += nth) {
      int b = i >> 8, tt = i & 255;
      const float* fv = A.fbuf + ((size_t)b * 256 + tt) * 8;
      const float* gv = A.gbuf + ((size_t)b * 256 + tt) * 8;
      float e = 0.0f;
      for (int d = 0; d < 128; ++d) {
        float s = A.Tb[d];
        for (int c = 0; c < 8; ++c) s += fv[c] * A.U[c * 128 + d] + gv[c] * A.Tw[c * 128 + d];
        e += A.vw[d] * tanhf(s);
      }
      e += A.prior[i];
      if (tt >= A.memlen[b]) e = -1e9f;
      A.ebuf[i] = e;
    }
    gsync(A.bar);

    // ---- softmax rows (wave per batch) + pa update + alignments out ----
    if (gw < 16) {
      int b = gw;
      const float* er = A.ebuf + b * 256;
      float mx = -3.0e38f;
      for (int tt = lane; tt < 256; tt += 32) mx = fmaxf(mx, er[tt]);
      for (int off = 16; off > 0; off >>= 1) mx = fmaxf(mx, __shfl_xor(mx, off, 32));
      float sum = 0.0f;
      for (int tt = lane; tt < 256; tt += 32) sum += __expf(er[tt] - mx);
      for (int off = 16; off > 0; off >>= 1) sum += __shfl_xor(sum, off, 32);
      float inv = 1.0f / sum;
      float* par = A.pa + b * 256;
      float* al = A.outAlign + ((size_t)b * 512 + t) * A.maxin;
      for (int tt = lane; tt < 256; tt += 32) {
        float av = __expf(er[tt] - mx) * inv;
        par[tt] = av;
        al[tt] = av;
      }
    }
    gsync(A.bar);

    // ---- context; build A2=[ah|ctx|dh] fragments + melFrag ctx part ----
    for (int i = tid; i < 16 * 768; i += nth) {
      int b = i / 768, d = i % 768;
      const float* par = A.pa + b * 256;
      const float* mem = A.memory + (size_t)b * 256 * 768 + d;
      float s = 0.0f;
      for (int tt = 0; tt < 256; ++tt) s += par[tt] * mem[(size_t)tt * 768];
      A.ctx[i] = s;
      __bf16 sb = hbf(s);
      A.aFrag[fragIndex(b, 1024 + d)] = sb;
      A.melFrag[fragIndex(b, 1024 + d)] = sb;
    }
    for (int i = tid; i < 16 * 1024; i += nth) {
      int b = i >> 10, j = i & 1023;
      A.aFrag[fragIndex(b, j)] = hbf(A.ah[i]);
      A.aFrag[fragIndex(b, 1792 + j)] = hbf(A.dh[i]);
    }
    gsync(A.bar);

    // ---- gate GEMM 2: A2 (K=2816, 88 tiles) @ [Wi;Wh] ----
    stageA(A.aFrag, sAbf, 5632);
    {
      const __bf16* bp = A.DWp + (size_t)gw * 88 * 512 + lane * 16;
      const __bf16* ap = sAbf + lane * 16;
      v8f a0 = {}, a1 = {};
      for (int kt = 0; kt < 88; kt += 2) {
        __builtin_prefetch(bp + (size_t)(kt + 2) * 512, 0, 1);
        v16bf x0 = *(const v16bf*)(ap + (size_t)kt * 512);
        v16bf w0 = *(const v16bf*)(bp + (size_t)kt * 512);
        a0 = wmma_bf16(x0, w0, a0);
        v16bf x1 = *(const v16bf*)(ap + (size_t)(kt + 1) * 512);
        v16bf w1 = *(const v16bf*)(bp + (size_t)(kt + 1) * 512);
        a1 = wmma_bf16(x1, w1, a1);
      }
      v8f acc = a0 + a1;
      int n = gw * 16 + (lane & 15);
      int m0 = (lane & 16) ? 8 : 0;
      float bb = A.db[n];
#pragma unroll
      for (int r = 0; r < 8; ++r) A.gates[(size_t)(m0 + r) * 4096 + n] = acc[r] + bb;
    }
    gsync(A.bar);

    // ---- dec LSTM pointwise + zoneout; emit melFrag dh part ----
    for (int i = tid; i < 16 * 1024; i += nth) {
      int b = i >> 10, j = i & 1023;
      const float* g = A.gates + (size_t)b * 4096;
      float ig = sigmf(g[j]), fg = sigmf(g[1024 + j]);
      float gg = tanhf(g[2048 + j]), og = sigmf(g[3072 + j]);
      float cn = fg * A.dc[i] + ig * gg;
      float hn = og * tanhf(cn);
      float nh = 0.1f * A.dh[i] + 0.9f * hn;
      A.dh[i] = nh;
      A.dc[i] = 0.1f * A.dc[i] + 0.9f * cn;
      A.melFrag[fragIndex(b, j)] = hbf(nh);
    }
    gsync(A.bar);

    // ---- mel = [dh|ctx]@Wm + bm via WMMA (waves 0..4); build next A1 ----
    if (gw < 5) {
      const __bf16* bp = A.WmP + (size_t)gw * 56 * 512 + lane * 16;
      const __bf16* ap = A.melFrag + lane * 16;
      v8f a0 = {}, a1 = {};
      for (int kt = 0; kt < 56; kt += 2) {
        v16bf x0 = *(const v16bf*)(ap + (size_t)kt * 512);
        v16bf w0 = *(const v16bf*)(bp + (size_t)kt * 512);
        a0 = wmma_bf16(x0, w0, a0);
        v16bf x1 = *(const v16bf*)(ap + (size_t)(kt + 1) * 512);
        v16bf w1 = *(const v16bf*)(bp + (size_t)(kt + 1) * 512);
        a1 = wmma_bf16(x1, w1, a1);
      }
      v8f acc = a0 + a1;
      int n = gw * 16 + (lane & 15);
      int b0 = (lane & 16) ? 8 : 0;
      float bb = A.bm[n];
#pragma unroll
      for (int r = 0; r < 8; ++r) {
        int b = b0 + r;
        float s = acc[r] + bb;
        size_t oi = ((size_t)b * 80 + n) * 512 + t;
        A.melbuf[oi] = s;
        A.outMel[oi] = (t >= A.outlen[b]) ? 0.0f : s;
      }
    }
    if (t + 1 < 512) {
      for (int i = tid; i < 16 * 2048; i += nth) {
        int b = i >> 11, c = i & 2047;
        float v;
        if (c < 256)       v = A.din[((size_t)(t + 1) * 16 + b) * 256 + c];
        else if (c < 1024) v = A.ctx[b * 768 + (c - 256)];
        else               v = A.ah[b * 1024 + (c - 1024)];
        A.aFrag[fragIndex(b, c)] = hbf(v);
      }
    }
    gsync(A.bar);
  }
}

// ---------------------------------------------------------------------------
// Postnet conv1d (K=5) as 5 shifted WMMA GEMMs. One block per (b, 16-t strip);
// the 5 shifted A-fragment sets are staged in LDS once and reused by all
// Cout tiles (waves iterate nt).
// ---------------------------------------------------------------------------
__global__ __launch_bounds__(256, 1) void k_conv_wmma(
    const float* __restrict__ X, const __bf16* __restrict__ Wp,
    const float* __restrict__ bias, float* __restrict__ Y,
    int Cin, int Ktiles, int Cout)
{
  extern __shared__ __align__(16) unsigned char smem[];
  __bf16* sA5 = (__bf16*)smem;                  // [5][Ktiles*512]
  int b = blockIdx.x >> 5;
  int tt = blockIdx.x & 31;
  int span = Ktiles * 512;
  int CinP = Ktiles * 32;
  // stage 5 shifted fragment sets (r fastest for coalescing)
  for (int i = threadIdx.x; i < 5 * 16 * CinP; i += blockDim.x) {
    int r = i & 15;
    int ci = (i >> 4) % CinP;
    int k = i / (16 * CinP);
    int tsrc = tt * 16 + r + k - 2;
    float v = (tsrc >= 0 && tsrc < 512 && ci < Cin)
                  ? X[((size_t)b * Cin + ci) * 512 + tsrc] : 0.0f;
    sA5[k * span + fragIndex(r, ci)] = hbf(v);
  }
  __syncthreads();

  int wib = threadIdx.x >> 5, lane = threadIdx.x & 31;
  int Ntiles = Cout >> 4;
  size_t perK = (size_t)Ktiles * Ntiles * 512;
  for (int nt = wib; nt < Ntiles; nt += 8) {
    v8f a0 = {}, a1 = {};
    for (int k = 0; k < 5; ++k) {
      const __bf16* ap = sA5 + k * span + lane * 16;
      const __bf16* bp = Wp + (size_t)k * perK + (size_t)nt * Ktiles * 512 + lane * 16;
      for (int kt = 0; kt < Ktiles; ++kt) {
        v16bf af = *(const v16bf*)(ap + (size_t)kt * 512);
        v16bf bf = *(const v16bf*)(bp + (size_t)kt * 512);
        if (kt & 1) a1 = wmma_bf16(af, bf, a1);
        else        a0 = wmma_bf16(af, bf, a0);
      }
    }
    v8f acc = a0 + a1;
    int n = nt * 16 + (lane & 15);
    int t0 = tt * 16 + ((lane & 16) ? 8 : 0);
    float bb = bias[n];
#pragma unroll
    for (int r = 0; r < 8; ++r)
      Y[((size_t)b * Cout + n) * 512 + t0 + r] = acc[r] + bb;
  }
}

// ---- batch-norm stats over (batch,time) per channel (Cout=512 layers) ----
__global__ void k_bn_stats(const float* __restrict__ Y, float* __restrict__ stats)
{
  __shared__ float ss[256], sq[256];
  int co = blockIdx.x;
  float s = 0.0f, q = 0.0f;
  for (int i = threadIdx.x; i < 8192; i += 256) {
    int b = i >> 9, t = i & 511;
    float v = Y[((size_t)b * 512 + co) * 512 + t];
    s += v; q += v * v;
  }
  ss[threadIdx.x] = s; sq[threadIdx.x] = q;
  __syncthreads();
  for (int off = 128; off > 0; off >>= 1) {
    if (threadIdx.x < off) {
      ss[threadIdx.x] += ss[threadIdx.x + off];
      sq[threadIdx.x] += sq[threadIdx.x + off];
    }
    __syncthreads();
  }
  if (threadIdx.x == 0) {
    float m = ss[0] * (1.0f / 8192.0f);
    float v = sq[0] * (1.0f / 8192.0f) - m * m;
    stats[co * 2] = m; stats[co * 2 + 1] = v;
  }
}

__global__ void k_bn_tanh(float* __restrict__ Y, const float* __restrict__ stats,
                          const float* __restrict__ gamma, const float* __restrict__ beta)
{
  int i = blockIdx.x * blockDim.x + threadIdx.x;
  if (i >= 16 * 512 * 512) return;
  int co = (i >> 9) & 511;
  float m = stats[co * 2], v = stats[co * 2 + 1];
  float x = (Y[i] - m) * rsqrtf(v + 1e-5f) * gamma[co] + beta[co];
  Y[i] = tanhf(x);
}

__global__ void k_post_final(const float* __restrict__ y5, const float* __restrict__ melbuf,
                             const int* __restrict__ outlen, float* __restrict__ out)
{
  int i = blockIdx.x * blockDim.x + threadIdx.x;
  if (i >= 16 * 80 * 512) return;
  int t = i & 511;
  int b = i / (80 * 512);
  float v = melbuf[i] + y5[i];
  if (t >= outlen[b]) v = 0.0f;
  out[i] = v;
}

// ---------------------------------------------------------------------------
// Host launcher
// ---------------------------------------------------------------------------
static inline int cdiv(long long a, int b) { return (int)((a + b - 1) / b); }

extern "C" void kernel_launch(void* const* d_in, const int* in_sizes, int n_in,
                              void* d_out, int out_size, void* d_ws, size_t ws_size,
                              hipStream_t stream)
{
  (void)in_sizes; (void)n_in; (void)ws_size;
  const float* x      = (const float*)d_in[0];
  const float* memory = (const float*)d_in[1];
  const int*   memlen = (const int*)d_in[2];
  const int*   outlen = (const int*)d_in[3];
  /* d_in[4] = max_input_len (derived from out_size instead) */
  const float* go = (const float*)d_in[5];
  const float* W1 = (const float*)d_in[6];
  const float* b1 = (const float*)d_in[7];
  const float* W2 = (const float*)d_in[8];
  const float* b2 = (const float*)d_in[9];
  const float* aWi = (const float*)d_in[10];
  const float* aWh = (const float*)d_in[11];
  const float* ab  = (const float*)d_in[12];
  const float* Wq = (const float*)d_in[13];
  const float* bq = (const float*)d_in[14];
  const float* V  = (const float*)d_in[15];
  const float* Fw = (const float*)d_in[16];
  const float* U  = (const float*)d_in[17];
  const float* Tw = (const float*)d_in[18];
  const float* Tb = (const float*)d_in[19];
  const float* vw = (const float*)d_in[20];
  const float* dWi = (const float*)d_in[21];
  const float* dWh = (const float*)d_in[22];
  const float* db  = (const float*)d_in[23];
  const float* Wm = (const float*)d_in[24];
  const float* bm = (const float*)d_in[25];
  const float* pw[5]; const float* pb[5]; const float* pg[4]; const float* pbe[4];
  {
    int idx = 26;
    for (int l = 0; l < 5; ++l) {
      pw[l] = (const float*)d_in[idx++];
      pb[l] = (const float*)d_in[idx++];
      if (l < 4) { pg[l] = (const float*)d_in[idx++]; pbe[l] = (const float*)d_in[idx++]; }
    }
  }

  float* outMel   = (float*)d_out;
  float* outMel2  = outMel + 655360;
  float* outAlign = outMel + 2 * 655360;
  int maxin = (out_size - 2 * 655360) / (16 * 512);

  // ---- workspace carve-up ----
  char* base = (char*)d_ws;
  size_t off = 0;
  auto alloc = [&](size_t bytes) -> void* {
    off = (off + 255) & ~(size_t)255;
    void* p = base + off;
    off += bytes;
    return p;
  };
  __bf16* W1p = (__bf16*)alloc((size_t)3 * 16 * 512 * 2);
  __bf16* W2p = (__bf16*)alloc((size_t)8 * 16 * 512 * 2);
  __bf16* AWp = (__bf16*)alloc((size_t)64 * 256 * 512 * 2);
  __bf16* DWp = (__bf16*)alloc((size_t)88 * 256 * 512 * 2);
  __bf16* WqP = (__bf16*)alloc((size_t)32 * 8 * 512 * 2);
  __bf16* WmP = (__bf16*)alloc((size_t)56 * 5 * 512 * 2);
  const int pCin[5] = {80, 512, 512, 512, 512};
  const int pCout[5] = {512, 512, 512, 512, 80};
  const int pKt[5] = {3, 16, 16, 16, 16};
  const int pNt[5] = {32, 32, 32, 32, 5};
  __bf16* PWp[5];
  for (int l = 0; l < 5; ++l)
    PWp[l] = (__bf16*)alloc((size_t)5 * pKt[l] * pNt[l] * 512 * 2);

  float* seq    = (float*)alloc((size_t)8192 * 96 * 4);
  float* h1     = (float*)alloc((size_t)8192 * 256 * 4);
  float* din    = (float*)alloc((size_t)8192 * 256 * 4);
  float* melbuf = (float*)alloc((size_t)655360 * 4);
  float* pA     = (float*)alloc((size_t)16 * 512 * 512 * 4);
  float* pB     = (float*)alloc((size_t)16 * 512 * 512 * 4);
  float* stats  = (float*)alloc(1024 * 4);
  __bf16* aFrag   = (__bf16*)alloc((size_t)88 * 512 * 2);
  __bf16* qFrag   = (__bf16*)alloc((size_t)32 * 512 * 2);
  __bf16* melFrag = (__bf16*)alloc((size_t)56 * 512 * 2);
  float* gates  = (float*)alloc((size_t)16 * 4096 * 4);
  float* ah = (float*)alloc(16384 * 4);
  float* ac = (float*)alloc(16384 * 4);
  float* dh = (float*)alloc(16384 * 4);
  float* dc = (float*)alloc(16384 * 4);
  float* pa    = (float*)alloc(4096 * 4);
  float* ebuf  = (float*)alloc(4096 * 4);
  float* prior = (float*)alloc(4096 * 4);
  float* q16   = (float*)alloc(2048 * 4);
  float* G     = (float*)alloc(2688 * 4);
  float* fbuf  = (float*)alloc(32768 * 4);
  float* gbuf  = (float*)alloc(32768 * 4);
  float* ctx   = (float*)alloc(12288 * 4);
  int*   bar   = (int*)alloc(256);

  // ---- init barrier + zero alignments region ----
  {
    size_t n = (size_t)16 * 512 * maxin;
    k_init<<<1024, 256, 0, stream>>>(bar, outAlign, n);
  }

  // ---- pack weights into WMMA fragment order ----
  auto pack = [&](const float* W, __bf16* outp, int Ntl, int KtT, int kt0, int ktC,
                  int sK, int sN, int sOff, int krow, int Ksrc, int Nsrc) {
    int total = ktC * Ntl * 32;
    k_pack_b<<<cdiv(total, 256), 256, 0, stream>>>(W, outp, Ntl, KtT, kt0, ktC,
                                                   sK, sN, sOff, krow, Ksrc, Nsrc);
  };
  pack(W1, W1p, 16, 3, 0, 3, 256, 1, 0, 0, 80, 256);
  pack(W2, W2p, 16, 8, 0, 8, 256, 1, 0, 0, 256, 256);
  pack(aWi, AWp, 256, 64, 0, 32, 4096, 1, 0, 0, 1024, 4096);
  pack(aWh, AWp, 256, 64, 32, 32, 4096, 1, 0, 1024, 1024, 4096);
  pack(dWi, DWp, 256, 88, 0, 56, 4096, 1, 0, 0, 1792, 4096);
  pack(dWh, DWp, 256, 88, 56, 32, 4096, 1, 0, 1792, 1024, 4096);
  pack(Wq, WqP, 8, 32, 0, 32, 128, 1, 0, 0, 1024, 128);
  pack(Wm, WmP, 5, 56, 0, 56, 80, 1, 0, 0, 1792, 80);
  for (int l = 0; l < 5; ++l) {
    size_t perK = (size_t)pKt[l] * pNt[l] * 512;
    for (int k = 0; k < 5; ++k)
      pack(pw[l], PWp[l] + (size_t)k * perK, pNt[l], pKt[l], 0, pKt[l],
           5, pCin[l] * 5, k, 0, pCin[l], pCout[l]);
  }

  // ---- prenet (parallel over all 512 steps) ----
  k_build_seq<<<cdiv(8192 * 96, 256), 256, 0, stream>>>(x, go, seq);
  k_gemm<<<cdiv((long long)512 * 16 * 32, 256), 256, 0, stream>>>(
      seq, W1p, b1, h1, 512, 16, 3, 96, 256, 1);
  k_gemm<<<cdiv((long long)512 * 16 * 32, 256), 256, 0, stream>>>(
      h1, W2p, b2, din, 512, 16, 8, 256, 256, 1);

  // ---- persistent recurrent decoder ----
  DecArgs da;
  da.din = din; da.memory = memory; da.memlen = memlen; da.outlen = outlen;
  da.AWp = AWp; da.DWp = DWp; da.WqP = WqP; da.WmP = WmP;
  da.ab = ab; da.db = db;
  da.bq = bq; da.V = V; da.Fw = Fw; da.U = U; da.Tw = Tw; da.Tb = Tb;
  da.vw = vw; da.bm = bm;
  da.aFrag = aFrag; da.qFrag = qFrag; da.melFrag = melFrag;
  da.gates = gates; da.ah = ah; da.ac = ac; da.dh = dh; da.dc = dc;
  da.pa = pa; da.ctx = ctx; da.q16 = q16; da.G = G;
  da.fbuf = fbuf; da.gbuf = gbuf; da.prior = prior; da.ebuf = ebuf;
  da.melbuf = melbuf; da.outMel = outMel; da.outAlign = outAlign;
  da.bar = bar; da.maxin = maxin;
  k_decoder<<<16, 512, 88 * 512 * 2, stream>>>(da);

  // ---- postnet ----
  auto conv = [&](const float* X, __bf16* Wp, const float* bias, float* Y,
                  int Cin, int Kt, int Cout) {
    k_conv_wmma<<<512, 256, (size_t)5 * Kt * 512 * 2, stream>>>(X, Wp, bias, Y,
                                                                Cin, Kt, Cout);
  };
  conv(melbuf, PWp[0], pb[0], pA, 80, 3, 512);
  k_bn_stats<<<512, 256, 0, stream>>>(pA, stats);
  k_bn_tanh<<<cdiv((long long)16 * 512 * 512, 256), 256, 0, stream>>>(pA, stats, pg[0], pbe[0]);
  conv(pA, PWp[1], pb[1], pB, 512, 16, 512);
  k_bn_stats<<<512, 256, 0, stream>>>(pB, stats);
  k_bn_tanh<<<cdiv((long long)16 * 512 * 512, 256), 256, 0, stream>>>(pB, stats, pg[1], pbe[1]);
  conv(pB, PWp[2], pb[2], pA, 512, 16, 512);
  k_bn_stats<<<512, 256, 0, stream>>>(pA, stats);
  k_bn_tanh<<<cdiv((long long)16 * 512 * 512, 256), 256, 0, stream>>>(pA, stats, pg[2], pbe[2]);
  conv(pA, PWp[3], pb[3], pB, 512, 16, 512);
  k_bn_stats<<<512, 256, 0, stream>>>(pB, stats);
  k_bn_tanh<<<cdiv((long long)16 * 512 * 512, 256), 256, 0, stream>>>(pB, stats, pg[3], pbe[3]);
  conv(pB, PWp[4], pb[4], pA, 512, 16, 80);
  k_post_final<<<cdiv(655360, 256), 256, 0, stream>>>(pA, melbuf, outlen, outMel2);
}
